// RelationMultiheadAttention_20959440404906
// MI455X (gfx1250) — matrix operand order repaired and checked by reading, not compiled, not run on hardware
//
#include <hip/hip_runtime.h>
#include <hip/hip_bf16.h>
#include <cstdint>

#define TGT 64
#define SRC 64
#define BSZ 128
#define EMB 64
#define NH  4
#define HD  16

typedef __attribute__((ext_vector_type(16))) __bf16 v16bf;
typedef __attribute__((ext_vector_type(8)))  float  v8f;
typedef __attribute__((ext_vector_type(4)))  unsigned int u32x4;
typedef __attribute__((ext_vector_type(8)))  int  i32x8;
typedef __attribute__((ext_vector_type(4)))  int  i32x4;

__device__ __forceinline__ unsigned short bfbits(float f) {   // RNE (one-time uses)
  union { float f; unsigned u; } a; a.f = f;
  unsigned r = a.u + 0x7FFFu + ((a.u >> 16) & 1u);
  return (unsigned short)(r >> 16);
}
__device__ __forceinline__ unsigned short bftrunc(float f) {  // fast truncating cast
  union { float f; unsigned u; } a; a.f = f;
  return (unsigned short)(a.u >> 16);
}
__device__ __forceinline__ __bf16 bits2bf(unsigned short s) {
  union { unsigned short u; __bf16 b; } o; o.u = s; return o.b;
}
__device__ __forceinline__ __bf16 f2bf(float f) { return bits2bf(bfbits(f)); }

// Inverse A-fragment map: (K, M) -> (lane, element), 16-bit A 16x32 layout:
// kr = b4*16 + hi*8 + gg*2 + lo.
__device__ __forceinline__ void akpos(int K, int M, int* lane, int* e) {
  int kr = K & 31;
  int hi = (kr >> 3) & 1;
  int b4 = (kr >> 4) & 1;
  int gg = (kr >> 1) & 3;
  *e = (b4 * 4 + gg) * 2 + (kr & 1);
  *lane = (M & 15) + hi * 16;
}

// Butterfly sum over the 16-lane DPP row via ROW_XMASK (VALU-only, no LDS).
__device__ __forceinline__ float row16_sum(float p) {
  int t;
  t = __builtin_amdgcn_update_dpp(0, __builtin_bit_cast(int, p), 0x168, 0xF, 0xF, true);
  p += __builtin_bit_cast(float, t);
  t = __builtin_amdgcn_update_dpp(0, __builtin_bit_cast(int, p), 0x164, 0xF, 0xF, true);
  p += __builtin_bit_cast(float, t);
  t = __builtin_amdgcn_update_dpp(0, __builtin_bit_cast(int, p), 0x162, 0xF, 0xF, true);
  p += __builtin_bit_cast(float, t);
  t = __builtin_amdgcn_update_dpp(0, __builtin_bit_cast(int, p), 0x161, 0xF, 0xF, true);
  p += __builtin_bit_cast(float, t);
  return p;
}

// ---------------------------------------------------------------------------
// Kernel 1: fused qkv projection.
// ---------------------------------------------------------------------------
__global__ void qkv_proj_kernel(const float* __restrict__ q_in,
                                const float* __restrict__ k_in,
                                const float* __restrict__ v_in,
                                const float* __restrict__ W,
                                const float* __restrict__ bias,
                                float* __restrict__ q_ws,
                                float* __restrict__ k_ws,
                                unsigned short* __restrict__ v_ws) {
  int idx = blockIdx.x * blockDim.x + threadIdx.x;
  if (idx >= TGT * BSZ * 3 * EMB) return;
  int row = idx / (3 * EMB);
  int c   = idx % (3 * EMB);
  const float* x  = (c < EMB) ? q_in : (c < 2 * EMB ? k_in : v_in);
  const float* xr = x + (size_t)row * EMB;
  const float* wr = W + (size_t)c * EMB;
  float acc = bias[c];
#pragma unroll
  for (int kk = 0; kk < EMB; ++kk) acc = fmaf(xr[kk], wr[kk], acc);
  if (c < EMB)            q_ws[(size_t)row * EMB + c] = acc;
  else if (c < 2 * EMB)   k_ws[(size_t)row * EMB + (c - EMB)] = acc;
  else                    v_ws[(size_t)row * EMB + (c - 2 * EMB)] = bfbits(acc);
}

// ---------------------------------------------------------------------------
// Kernel 2: fused relation-projection + attention + out-projection with
// double-buffered TDM streaming of the relation tensor.
// ---------------------------------------------------------------------------
struct ChunkBufs {                      // chunk-loop phase
  __align__(16) float relF32[2][32][64];                 // 16 KB, TDM landing x2
  __align__(32) unsigned short fragA[2][2][2][32][16];   // 8 KB, bf16 A-frags x2
};
struct PostBufs {                       // post phase (aliases chunk bufs)
  __align__(32) unsigned short fragP[NH][2][32][16];     // 8 KB prob A-frags
  __align__(32) unsigned short fragO[2][32][16];         // 2 KB head-merge frags
};
union ShUnion { ChunkBufs c; PostBufs p; };

__global__ void __launch_bounds__(128)
rel_attention_kernel(const float* __restrict__ q_ws,
                     const float* __restrict__ k_ws,
                     const unsigned short* __restrict__ v_ws,
                     const float* __restrict__ relation,
                     const float* __restrict__ rel_weight,
                     const float* __restrict__ rel_bias,
                     const float* __restrict__ out_weight,
                     const float* __restrict__ out_bias,
                     float* __restrict__ out) {
  __shared__ float scores[NH][16][SRC];   // 16 KB
  __shared__ ShUnion sh;                  // 24 KB

  const int tid = threadIdx.x;
  const int h   = tid >> 5;        // head == wave id
  const int L   = tid & 31;        // lane
  const int n   = L & 15;
  const int hiL = L >> 4;
  const int i0  = blockIdx.x * 16;
  const int b   = blockIdx.y;

  const float biasA = rel_bias[h * 16 + n];
  const float biasB = rel_bias[EMB + h * 16 + n];

  // QA = (q + biasA)/16, statically indexed (il = m*8 + 4*hiL + t).
  float QA[8];
#pragma unroll
  for (int mi = 0; mi < 2; ++mi)
#pragma unroll
    for (int t = 0; t < 4; ++t)
      QA[mi * 4 + t] = 0.0625f *
          (q_ws[((size_t)(i0 + mi * 8 + 4 * hiL + t) * BSZ + b) * EMB + h * 16 + n]
           + biasA);

  // Per-lane B fragments of rel_weight (this head's ra/rb column slices), RNE.
  v16bf Bra[2], Brb[2];
#pragma unroll
  for (int kf = 0; kf < 2; ++kf) {
#pragma unroll
    for (int e = 0; e < 16; ++e) {
      Bra[kf][e] = f2bf(rel_weight[(size_t)(h * 16 + n) * EMB + kf * 32 + 8 * hiL + (e & 7) + ((e >> 3) << 4)]);
      Brb[kf][e] = f2bf(rel_weight[(size_t)(EMB + h * 16 + n) * EMB + kf * 32 + 8 * hiL + (e & 7) + ((e >> 3) << 4)]);
    }
  }

  // TDM descriptor invariants: 3D tile x=64 f32, y=2 (stride BSZ*EMB),
  // z=16 (stride SRC*BSZ*EMB via dim1_stride).
  i32x8 g1;
  g1[0] = 0x00020000;                 // data_size=4B, no multicast
  g1[1] = (64 << 16);                 // tensor_dim0 = 64
  g1[2] = (int)0xFFFF0000u;           // tensor_dim1 = 0xFFFF
  g1[3] = (64 << 16);                 // tile_dim0 = 64
  g1[4] = 2 | (16 << 16);             // tile_dim1 = 2, tile_dim2 = 16
  g1[5] = BSZ * EMB;                  // dim0_stride = 8192
  g1[6] = (int)(((SRC * BSZ * EMB) & 0xFFFF) << 16);
  g1[7] = (SRC * BSZ * EMB) >> 16;    // dim1_stride = 524288
  i32x4 g2; g2[0] = 0xFFFF; g2[1] = 0; g2[2] = 0; g2[3] = 0;
  i32x4 g3; g3[0] = 0; g3[1] = 0; g3[2] = 0; g3[3] = 0;
  i32x8 g4;
  g4[0] = 0; g4[1] = 0; g4[2] = 0; g4[3] = 0;
  g4[4] = 0; g4[5] = 0; g4[6] = 0; g4[7] = 0;

  auto tdm_issue = [&](int jcN, int buf) {
    unsigned long long gaddr = (unsigned long long)(uintptr_t)(relation +
        ((((size_t)i0) * SRC + jcN * 2) * BSZ + b) * EMB);
    u32x4 g0;
    g0[0] = 1u;
    g0[1] = (unsigned)(uintptr_t)(&sh.c.relF32[buf][0][0]);
    g0[2] = (unsigned)(gaddr & 0xFFFFFFFFu);
    g0[3] = (unsigned)((gaddr >> 32) & 0x1FFFFFFu) | (2u << 30);
    __builtin_amdgcn_tensor_load_to_lds(g0, g1, g2, g3, g4, 0);
  };

  if (h == 0) tdm_issue(0, 0);        // prime the pipeline

  for (int jc = 0; jc < 32; ++jc) {
    const int cur = jc & 1;
    if (h == 0) {
      if (jc + 1 < 32) {
        tdm_issue(jc + 1, cur ^ 1);   // overlap next DMA with this chunk's work
        __builtin_amdgcn_s_wait_tensorcnt(1);   // oldest (cur) complete
      } else {
        __builtin_amdgcn_s_wait_tensorcnt(0);
      }
    }
    __syncthreads();                  // relF32[cur] ready; fragA[cur] free

    // f32 -> bf16 (truncating, v_perm packs 2) into A-fragment order.
    // Quad of consecutive K lands in 4 consecutive fragment elements (8B store).
    for (int x = tid; x < 32 * 16; x += 128) {
      int r  = x >> 4;                // chunk row (= i_local*2 + j_local)
      int K0 = (x & 15) * 4;
      uint4 u = *(const uint4*)&sh.c.relF32[cur][r][K0];
      unsigned p0 = __builtin_amdgcn_perm(u.y, u.x, 0x07060302u);
      unsigned p1 = __builtin_amdgcn_perm(u.w, u.z, 0x07060302u);
      int lane, e;
      akpos(K0, r & 15, &lane, &e);
      uint2 d; d.x = p0; d.y = p1;
      *(uint2*)&sh.c.fragA[cur][r >> 4][K0 >> 5][lane][e] = d;
    }
    __syncthreads();

    // k rows for this chunk; KB = k + biasB hoisted.
    float KB0 = k_ws[((size_t)(jc * 2 + 0) * BSZ + b) * EMB + h * 16 + n] + biasB;
    float KB1 = k_ws[((size_t)(jc * 2 + 1) * BSZ + b) * EMB + h * 16 + n] + biasB;

#pragma unroll
    for (int m = 0; m < 2; ++m) {
      v8f cra = {0, 0, 0, 0, 0, 0, 0, 0};
      v8f crb = {0, 0, 0, 0, 0, 0, 0, 0};
#pragma unroll
      for (int kf = 0; kf < 2; ++kf) {
        v16bf A = *(const v16bf*)&sh.c.fragA[cur][m][kf][L][0];
        cra = __builtin_amdgcn_wmma_f32_16x16x32_bf16(false, A, false, Bra[kf],
                                                      (short)0, cra, false, false);
        crb = __builtin_amdgcn_wmma_f32_16x16x32_bf16(false, A, false, Brb[kf],
                                                      (short)0, crb, false, false);
      }
      // score = (QA + cra/16) . (KB + crb), reduced over hd via DPP xmask.
      float keep = 0.f;
#pragma unroll
      for (int rr = 0; rr < 8; ++rr) {
        float pa = fmaf(cra[rr], 0.0625f, QA[m * 4 + (rr >> 1)]);
        float pb = ((rr & 1) ? KB1 : KB0) + crb[rr];
        float p = row16_sum(pa * pb);
        keep = (n == rr) ? p : keep;   // v_cndmask keeper
      }
      if (n < 8) {
        int rowc = m * 16 + n + 8 * hiL;
        scores[h][rowc >> 1][jc * 2 + (n & 1)] = keep;
      }
    }
  }
  __syncthreads();

  // Softmax over j; write probs in A-fragment order (bf16, truncating).
  if (tid < NH * 16) {
    int hh = tid >> 4, il = tid & 15;
    float mx = -3.0e38f;
    for (int j = 0; j < SRC; ++j) mx = fmaxf(mx, scores[hh][il][j]);
    float sum = 0.f;
    for (int j = 0; j < SRC; ++j) {
      float e = __expf(scores[hh][il][j] - mx);
      scores[hh][il][j] = e;
      sum += e;
    }
    float inv = 1.0f / sum;
    for (int j = 0; j < SRC; ++j) {
      int lane, e;
      akpos(j, il, &lane, &e);
      sh.p.fragP[hh][j >> 5][lane][e] = bftrunc(scores[hh][il][j] * inv);
    }
  }
  __syncthreads();

  // attn @ v per head: probs(16x64) x v(64x16) -> 2 WMMAs.
  {
    v8f co = {0, 0, 0, 0, 0, 0, 0, 0};
#pragma unroll
    for (int kf = 0; kf < 2; ++kf) {
      v16bf A = *(const v16bf*)&sh.p.fragP[h][kf][L][0];
      v16bf Bv;
#pragma unroll
      for (int e = 0; e < 16; ++e)
        Bv[e] = bits2bf(v_ws[((size_t)(kf * 32 + 8 * hiL + (e & 7) + ((e >> 3) << 4)) * BSZ + b) * EMB + h * 16 + n]);
      co = __builtin_amdgcn_wmma_f32_16x16x32_bf16(false, A, false, Bv,
                                                   (short)0, co, false, false);
    }
#pragma unroll
    for (int rr = 0; rr < 8; ++rr) {
      int M = rr + 8 * hiL;
      int c = h * 16 + n;
      int lane, e;
      akpos(c, M, &lane, &e);
      sh.p.fragO[c >> 5][lane][e] = bfbits(co[rr]);
    }
  }
  __syncthreads();

  // Fused output projection: wave h computes output columns [h*16, h*16+16).
  {
    v8f cf = {0, 0, 0, 0, 0, 0, 0, 0};
#pragma unroll
    for (int kf = 0; kf < 2; ++kf) {
      v16bf A = *(const v16bf*)&sh.p.fragO[kf][L][0];
      v16bf Bw;
#pragma unroll
      for (int e = 0; e < 16; ++e)
        Bw[e] = f2bf(out_weight[(size_t)(h * 16 + n) * EMB + kf * 32 + 8 * hiL + (e & 7) + ((e >> 3) << 4)]);
      cf = __builtin_amdgcn_wmma_f32_16x16x32_bf16(false, A, false, Bw,
                                                   (short)0, cf, false, false);
    }
    float ob = out_bias[h * 16 + n];
#pragma unroll
    for (int rr = 0; rr < 8; ++rr) {
      int i = i0 + rr + 8 * hiL;
      out[((size_t)i * BSZ + b) * EMB + h * 16 + n] = cf[rr] + ob;
    }
  }
}

extern "C" void kernel_launch(void* const* d_in, const int* in_sizes, int n_in,
                              void* d_out, int out_size, void* d_ws, size_t ws_size,
                              hipStream_t stream) {
  (void)in_sizes; (void)n_in; (void)out_size; (void)ws_size;
  const float* query    = (const float*)d_in[0];
  const float* key_     = (const float*)d_in[1];
  const float* value    = (const float*)d_in[2];
  const float* relation = (const float*)d_in[3];
  const float* in_w     = (const float*)d_in[4];
  const float* in_b     = (const float*)d_in[5];
  const float* rel_w    = (const float*)d_in[6];
  const float* rel_b    = (const float*)d_in[7];
  const float* out_w    = (const float*)d_in[8];
  const float* out_b    = (const float*)d_in[9];
  float* out = (float*)d_out;

  float* q_ws = (float*)d_ws;                                   // 2 MB
  float* k_ws = q_ws + (size_t)TGT * BSZ * EMB;                 // 2 MB
  unsigned short* v_ws = (unsigned short*)(k_ws + (size_t)SRC * BSZ * EMB); // 1 MB

  int total = TGT * BSZ * 3 * EMB;
  qkv_proj_kernel<<<(total + 255) / 256, 256, 0, stream>>>(
      query, key_, value, in_w, in_b, q_ws, k_ws, v_ws);

  dim3 grid(TGT / 16, BSZ);
  rel_attention_kernel<<<grid, 128, 0, stream>>>(
      q_ws, k_ws, v_ws, relation, rel_w, rel_b, out_w, out_b, out);
}
